// MultiheadAttention_17368847745196
// MI455X (gfx1250) — matrix operand order repaired and checked by reading, not compiled, not run on hardware
//
#include <hip/hip_runtime.h>

typedef __bf16 bf16;
typedef __attribute__((ext_vector_type(16))) __bf16 v16bf;
typedef __attribute__((ext_vector_type(8)))  float  v8f;

#define NEGV -9.0e15f

union FragU { v16bf v; uint4 u[2]; };

// A-fragment: 16(M) x 32(K) bf16. lane&15 = M row; lanes 0-15 hold K {0..7,16..23},
// lanes 16-31 hold K {8..15,24..31} (ISA 16-bit A layout).
__device__ inline v16bf load_fragA(const bf16* p, int ld) {
  const int lane = threadIdx.x;
  const int r  = lane & 15;
  const int kb = (lane >> 4) << 3;              // 0 or 8
  const bf16* q = p + r * ld + kb;
  FragU f;
  f.u[0] = *(const uint4*)(q);                  // K = kb .. kb+7
  f.u[1] = *(const uint4*)(q + 16);             // K = kb+16 .. kb+23
  return f.v;
}

// B-fragment: 32(K) x 16(N) bf16, loaded from row-major B^T (N x K, ld = row stride).
// lane&15 = N column; lanes 0-15 hold K {0..7,16..23}, lanes 16-31 hold K {8..15,24..31}.
__device__ inline v16bf load_fragB(const bf16* p, int ld) {
  const int lane = threadIdx.x;
  const int n  = lane & 15;
  const int kb = (lane >> 4) << 3;              // 0 or 8
  const bf16* q = p + n * ld + kb;
  FragU f;
  f.u[0] = *(const uint4*)(q);                  // K = kb .. kb+7
  f.u[1] = *(const uint4*)(q + 16);             // K = kb+16 .. kb+23
  return f.v;
}

__device__ inline v8f wmma_bf16(v16bf a, v16bf b, v8f c) {
  return __builtin_amdgcn_wmma_f32_16x16x32_bf16(false, a, false, b, (short)0, c,
                                                 false, false);
}

// ---------------------------------------------------------------- convert
__global__ void cvt_f32_bf16(const float* __restrict__ src, bf16* __restrict__ dst, int n) {
  int i = blockIdx.x * blockDim.x + threadIdx.x;
  if (i < n) dst[i] = (bf16)src[i];
}

// ---------------------------------------------------------------- QKV GEMM
// out[m, e] = sum_k X[m,k] * Wqkv[e,k] + b[e];  M=8192, N=1536, K=512.
// e = h*192 + which*64 + dh. Q,K stored (b,h,s,dh); V stored transposed (b,h,dh,s).
__global__ void qkv_gemm(const bf16* __restrict__ X, const bf16* __restrict__ W,
                         const float* __restrict__ bias,
                         bf16* __restrict__ Q, bf16* __restrict__ Kd,
                         bf16* __restrict__ Vt) {
  const int lane = threadIdx.x;
  const int m0 = (blockIdx.x * 8 + threadIdx.y) * 16;
  const int n0 = blockIdx.y * 64;
  v8f acc[4] = {};
  for (int k0 = 0; k0 < 512; k0 += 32) {
    v16bf a = load_fragA(X + (size_t)m0 * 512 + k0, 512);
#pragma unroll
    for (int t = 0; t < 4; ++t) {
      v16bf bf = load_fragB(W + (size_t)(n0 + 16 * t) * 512 + k0, 512);
      acc[t] = wmma_bf16(a, bf, acc[t]);
    }
  }
  const int h = n0 / 192;
  const int which = (n0 % 192) / 64;
  const int lg = lane >> 4, c = lane & 15;
#pragma unroll
  for (int t = 0; t < 4; ++t) {
    const int dh = 16 * t + c;
    const float bv = bias[n0 + 16 * t + c];
#pragma unroll
    for (int i = 0; i < 8; ++i) {
      const int m = m0 + i + 8 * lg;
      const int b = m >> 11, s = m & 2047;
      const bf16 val = (bf16)(acc[t][i] + bv);
      if (which == 0)      Q [((size_t)(b * 8 + h) * 2048 + s) * 64 + dh] = val;
      else if (which == 1) Kd[((size_t)(b * 8 + h) * 2048 + s) * 64 + dh] = val;
      else                 Vt[((size_t)(b * 8 + h) * 64 + dh) * 2048 + s] = val;
    }
  }
}

// ---------------------------------------------------------------- fused flash attention
// grid (S/64, B*H), block (32,4): each wave = 16 query rows, full DH=64.
__global__ void attn_kernel(const bf16* __restrict__ Q, const bf16* __restrict__ K,
                            const bf16* __restrict__ Vt, const int* __restrict__ mask,
                            bf16* __restrict__ O) {
  __shared__ __align__(16) bf16 lds[4][16 * 32];
  const int lane = threadIdx.x;
  const int wv = threadIdx.y;
  const int bh = blockIdx.y;
  const int b = bh >> 3, h = bh & 7;
  const int q0 = blockIdx.x * 64 + wv * 16;
  const bf16* Qbh = Q  + (size_t)bh * 2048 * 64;
  const bf16* Kbh = K  + (size_t)bh * 2048 * 64;
  const bf16* Vbh = Vt + (size_t)bh * 64 * 2048;
  const int* mrow = mask + b * 2048;
  const int lg = lane >> 4, c = lane & 15;

  const v16bf qf0 = load_fragA(Qbh + (size_t)q0 * 64 + 0, 64);
  const v16bf qf1 = load_fragA(Qbh + (size_t)q0 * 64 + 32, 64);

  float m_run[8], l_run[8];
  v8f accO[4] = {};
#pragma unroll
  for (int i = 0; i < 8; ++i) { m_run[i] = NEGV; l_run[i] = 0.f; }

  for (int kb = 0; kb < 2048; kb += 32) {
    v8f lg0 = {}, lg1 = {};
    {
      v16bf kf;
      kf = load_fragB(Kbh + (size_t)(kb + 0)  * 64 + 0,  64); lg0 = wmma_bf16(qf0, kf, lg0);
      kf = load_fragB(Kbh + (size_t)(kb + 0)  * 64 + 32, 64); lg0 = wmma_bf16(qf1, kf, lg0);
      kf = load_fragB(Kbh + (size_t)(kb + 16) * 64 + 0,  64); lg1 = wmma_bf16(qf0, kf, lg1);
      kf = load_fragB(Kbh + (size_t)(kb + 16) * 64 + 32, 64); lg1 = wmma_bf16(qf1, kf, lg1);
    }
    const int v0 = mrow[kb + c];
    const int v1 = mrow[kb + 16 + c];
#pragma unroll
    for (int i = 0; i < 8; ++i) {
      lg0[i] = v0 ? lg0[i] * 0.125f : NEGV;
      lg1[i] = v1 ? lg1[i] * 0.125f : NEGV;
    }
    float mnew[8], alpha[8];
#pragma unroll
    for (int i = 0; i < 8; ++i) {
      float r = fmaxf(lg0[i], lg1[i]);
#pragma unroll
      for (int off = 1; off < 16; off <<= 1) r = fmaxf(r, __shfl_xor(r, off, 32));
      mnew[i]  = fmaxf(m_run[i], r);
      alpha[i] = __expf(m_run[i] - mnew[i]);
      m_run[i] = mnew[i];
    }
#pragma unroll
    for (int i = 0; i < 8; ++i) {
      const float p0 = __expf(lg0[i] - mnew[i]);
      const float p1 = __expf(lg1[i] - mnew[i]);
      float rs = p0 + p1;
#pragma unroll
      for (int off = 1; off < 16; off <<= 1) rs += __shfl_xor(rs, off, 32);
      l_run[i] = l_run[i] * alpha[i] + rs;
      const int row = i + 8 * lg;
      lds[wv][row * 32 + c]      = (bf16)p0;
      lds[wv][row * 32 + 16 + c] = (bf16)p1;
    }
    asm volatile("s_wait_dscnt 0" ::: "memory");   // P tile visible before re-read
    const v16bf pf = load_fragA(&lds[wv][0], 32);
#pragma unroll
    for (int t = 0; t < 4; ++t) {
#pragma unroll
      for (int i = 0; i < 8; ++i) accO[t][i] *= alpha[i];
      v16bf vf = load_fragB(Vbh + (size_t)(16 * t) * 2048 + kb, 2048);
      accO[t] = wmma_bf16(pf, vf, accO[t]);
    }
  }
#pragma unroll
  for (int i = 0; i < 8; ++i) l_run[i] = 1.0f / l_run[i];
#pragma unroll
  for (int t = 0; t < 4; ++t) {
    const int dh = 16 * t + c;
#pragma unroll
    for (int i = 0; i < 8; ++i) {
      const int s = q0 + i + 8 * lg;
      O[((size_t)(b * 2048 + s)) * 512 + h * 64 + dh] = (bf16)(accO[t][i] * l_run[i]);
    }
  }
}

// ---------------------------------------------------------------- output GEMM
__global__ void out_gemm(const bf16* __restrict__ X, const bf16* __restrict__ W,
                         const float* __restrict__ bias, float* __restrict__ Out) {
  const int lane = threadIdx.x;
  const int m0 = (blockIdx.x * 8 + threadIdx.y) * 16;
  const int n0 = blockIdx.y * 64;
  v8f acc[4] = {};
  for (int k0 = 0; k0 < 512; k0 += 32) {
    v16bf a = load_fragA(X + (size_t)m0 * 512 + k0, 512);
#pragma unroll
    for (int t = 0; t < 4; ++t) {
      v16bf bf = load_fragB(W + (size_t)(n0 + 16 * t) * 512 + k0, 512);
      acc[t] = wmma_bf16(a, bf, acc[t]);
    }
  }
  const int lg = lane >> 4, c = lane & 15;
#pragma unroll
  for (int t = 0; t < 4; ++t) {
    const int n = n0 + 16 * t + c;
    const float bv = bias[n];
#pragma unroll
    for (int i = 0; i < 8; ++i) {
      const int m = m0 + i + 8 * lg;
      Out[(size_t)m * 512 + n] = acc[t][i] + bv;
    }
  }
}

// ---------------------------------------------------------------- launch
extern "C" void kernel_launch(void* const* d_in, const int* in_sizes, int n_in,
                              void* d_out, int out_size, void* d_ws, size_t ws_size,
                              hipStream_t stream) {
  const float* x    = (const float*)d_in[0];
  const int*   mask = (const int*)d_in[1];
  const float* Wqkv = (const float*)d_in[2];
  const float* bqkv = (const float*)d_in[3];
  const float* Wo   = (const float*)d_in[4];
  const float* bo   = (const float*)d_in[5];
  float* out = (float*)d_out;

  char* ws = (char*)d_ws;
  bf16* xb    = (bf16*)(ws);                 // 8192x512           (8.39 MB)
  bf16* wqkvb = (bf16*)(ws +  8388608);      // 1536x512           (1.57 MB)
  bf16* wob   = (bf16*)(ws +  9961472);      // 512x512            (0.52 MB)
  bf16* qws   = (bf16*)(ws + 10485760);      // (B*H,S,64)         (8.39 MB)
  bf16* kws   = (bf16*)(ws + 18874368);      // (B*H,S,64)         (8.39 MB)
  bf16* vtws  = (bf16*)(ws + 27262976);      // (B*H,64,S)         (8.39 MB)
  bf16* vals  = (bf16*)(ws + 35651584);      // (B*S,512)          (8.39 MB)

  cvt_f32_bf16<<<(4194304 + 255) / 256, 256, 0, stream>>>(x, xb, 4194304);
  cvt_f32_bf16<<<(786432  + 255) / 256, 256, 0, stream>>>(Wqkv, wqkvb, 786432);
  cvt_f32_bf16<<<(262144  + 255) / 256, 256, 0, stream>>>(Wo, wob, 262144);

  qkv_gemm  <<<dim3(64, 24), dim3(32, 8), 0, stream>>>(xb, wqkvb, bqkv, qws, kws, vtws);
  attn_kernel<<<dim3(32, 32), dim3(32, 4), 0, stream>>>(qws, kws, vtws, mask, vals);
  out_gemm  <<<dim3(64, 8),  dim3(32, 8), 0, stream>>>(vals, wob, bo, out);
}
